// G3DGraphConvolution_59545426592336
// MI455X (gfx1250) — matrix operand order repaired and checked by reading, not compiled, not run on hardware
//
#include <hip/hip_runtime.h>
#include <hip/hip_bf16.h>
#include <stdint.h>

#define B_    32
#define CIN   96
#define T_    288
#define V_    25
#define S_    3
#define COUT  96
#define M_    (S_*COUT)     // 288 rows of stage-1 GEMM
#define K_    CIN           // 96
#define TT    4             // t-values per workgroup
#define NN    112           // padded N columns (TT*V_=100 -> 112)
#define NT_M  18            // 288/16
#define NT_N  7             // 112/16
#define NTILES (NT_M*NT_N)  // 126
#define K2    96            // stage-2 K = S_*V_ = 75 padded to 96
#define NT2   48            // stage-2 tiles: 4 tl * 6 c-tiles * 2 m-tiles

typedef __attribute__((ext_vector_type(16))) __bf16 v16bf;
typedef __attribute__((ext_vector_type(8)))  float  v8f;

union Frag { v16bf v; uint4 q[2]; };

// CDNA5 async global->LDS copy (ASYNCcnt) if the toolchain exposes it
#if defined(__has_builtin)
#  if __has_builtin(__builtin_amdgcn_global_load_async_to_lds_b128)
#    define HAVE_ASYNC_LDS 1
#  endif
#endif

#ifdef HAVE_ASYNC_LDS
typedef int vint4 __attribute__((vector_size(16)));
typedef __attribute__((address_space(1))) vint4 g_vint4;   // global
typedef __attribute__((address_space(3))) vint4 l_vint4;   // LDS
#endif

// workspace layout (bytes)
#define WS_WBF16  0                  // 27648 bf16 conv_w  (55296 B)
#define WS_SUM    55296              // 96 f32
#define WS_SUMSQ  (55296 + 384)     // 96 f32
#define WS_SCALE  (55296 + 768)     // 96 f32
#define WS_SHIFT  (55296 + 1152)    // 96 f32

// LDS layout for fused kernel (bytes) -- total 157,824 B -> 2 workgroups/WGP
#define L_WA   0        // bf16[288][96]     = 55296   stage-1 A operand (weights)
#define L_XB   55296    // bf16[112][96]     = 21504   stage-1 B operand (x tile)
#define L_CB   76800    // f32[288]          = 1152    bias
#define L_AB   77952    // bf16[32][96]      = 6144    stage-2 B operand (adjacency^T)
#define L_YB   84096    // bf16[4][96][96]   = 73728   y in stage-2 A-operand layout
#define L_TOTAL (84096 + TT*COUT*K2*2)

// ---------------------------------------------------------------------------
// Kernel A: convert conv_w fp32 -> bf16 (row-major [o][k]) once; zero stats.
// ---------------------------------------------------------------------------
__global__ __launch_bounds__(256) void prep_kernel(const float* __restrict__ conv_w,
                                                   unsigned char* __restrict__ ws) {
    int gid = blockIdx.x * 256 + threadIdx.x;
    uint32_t* wout = (uint32_t*)(ws + WS_WBF16);
    if (gid < (M_ * K_) / 2) {
        union { __bf16 h[2]; uint32_t u; } p;
        p.h[0] = (__bf16)conv_w[gid * 2];
        p.h[1] = (__bf16)conv_w[gid * 2 + 1];
        wout[gid] = p.u;
    }
    if (blockIdx.x == 0 && threadIdx.x < 192) {           // zero sum+sumsq
        ((float*)(ws + WS_SUM))[threadIdx.x] = 0.0f;
    }
}

// ---------------------------------------------------------------------------
// Kernel B: fully WMMA-fused  y = W·x + b  then  z = y ⊗ A, both on the
// matrix pipe.  One workgroup = one (batch, 4-t slab).
// ---------------------------------------------------------------------------
__global__ __launch_bounds__(256) void conv_graph_kernel(
    const float* __restrict__ x, const float* __restrict__ Aadj,
    const float* __restrict__ conv_b, const unsigned char* __restrict__ ws,
    float* __restrict__ z)
{
    extern __shared__ __align__(16) unsigned char smem[];
    __bf16* wA = (__bf16*)(smem + L_WA);
    __bf16* xb = (__bf16*)(smem + L_XB);
    float*  cb = (float* )(smem + L_CB);
    __bf16* ab = (__bf16*)(smem + L_AB);
    __bf16* yb = (__bf16*)(smem + L_YB);

    const int tid = threadIdx.x;
    const int b   = blockIdx.x / (T_ / TT);
    const int t0  = (blockIdx.x % (T_ / TT)) * TT;

    // ---- stage weights / bias / adjacency / x-tile into LDS ----
    {
#ifdef HAVE_ASYNC_LDS
        // async DMA the bf16 weights (pure copy) while VALU converts below
        for (int i = tid; i < (M_ * K_ * 2) / 16; i += 256) {
            __builtin_amdgcn_global_load_async_to_lds_b128(
                (g_vint4*)(uintptr_t)(ws + WS_WBF16 + i * 16),
                (l_vint4*)(uintptr_t)(smem + L_WA + i * 16),
                0, 0);
        }
#else
        const uint32_t* wsrc = (const uint32_t*)(ws + WS_WBF16);
        uint32_t* wdst = (uint32_t*)wA;
        for (int i = tid; i < (M_ * K_) / 2; i += 256) wdst[i] = wsrc[i];
#endif
        for (int i = tid; i < M_; i += 256) cb[i] = conv_b[i];
        // zero y buffer (covers K/N padding for stage 2)
        uint32_t* yz = (uint32_t*)yb;
        for (int i = tid; i < (TT * COUT * K2) / 2; i += 256) yz[i] = 0u;
        // adjacency -> bf16, transposed to [m][k=(s,n)], zero padded
        for (int i = tid; i < 32 * K2; i += 256) {
            int m = i / K2, k = i % K2;
            float v = 0.0f;
            if (m < V_ && k < S_ * V_) {
                int s = k / V_, n = k % V_;
                v = Aadj[(s * V_ + n) * V_ + m];
            }
            ab[i] = (__bf16)v;
        }
        // x tile, transposed to [n][k] so each lane's K-run is contiguous
        for (int i = tid; i < K_ * NN; i += 256) {
            int k = i / NN, n = i % NN;
            float v = 0.0f;
            if (n < TT * V_) {
                int t = t0 + n / V_, vv = n % V_;
                v = x[((size_t)(b * CIN + k) * T_ + t) * V_ + vv];
            }
            xb[n * K_ + k] = (__bf16)v;
        }
    }
#ifdef HAVE_ASYNC_LDS
#  if __has_builtin(__builtin_amdgcn_s_wait_asynccnt)
    __builtin_amdgcn_s_wait_asynccnt(0);
#  else
    asm volatile("s_wait_asynccnt 0x0" ::: "memory");
#  endif
#endif
    __syncthreads();

    const int wave = tid >> 5;
    const int lane = tid & 31;
    const int hlf  = lane >> 4;     // lane-half selects K sub-block per ISA layout
    const int ln   = lane & 15;

    // ---- stage 1: 126 16x16 tiles, K=96 = 3 x wmma_f32_16x16x32_bf16 ----
    for (int tile = wave; tile < NTILES; tile += 8) {
        const int mt = tile / NT_N, nt = tile % NT_N;
        const __bf16* aBase = wA + (mt * 16 + ln) * K_ + hlf * 8;
        const __bf16* bBase = xb + (nt * 16 + ln) * K_ + hlf * 8;
        Frag af[3], bfr[3];
#pragma unroll
        for (int kk = 0; kk < 3; ++kk) {            // hoisted: 12 ds_load_b128
            af[kk].q[0]  = *(const uint4*)(aBase + kk * 32);
            af[kk].q[1]  = *(const uint4*)(aBase + kk * 32 + 16);
            bfr[kk].q[0] = *(const uint4*)(bBase + kk * 32);
            bfr[kk].q[1] = *(const uint4*)(bBase + kk * 32 + 16);
        }
        v8f acc = {};
#pragma unroll
        for (int kk = 0; kk < 3; ++kk)
            acc = __builtin_amdgcn_wmma_f32_16x16x32_bf16(
                false, af[kk].v, false, bfr[kk].v, (short)0, acc, false, false);

        // D: VGPR r -> (o = obase + r, n0 = nt*16 + ln).  obase is a multiple
        // of 8 and 96 % 8 == 0, so s and cbase are invariant over r: hoist.
        const int n0 = nt * 16 + ln;
        if (n0 < TT * V_) {
            const int tl = n0 / V_, vv = n0 % V_;
            const int obase = mt * 16 + hlf * 8;
            const int s     = obase / COUT;
            const int cbase = obase - s * COUT;
            const float* cbp = cb + obase;
            __bf16* ybp = yb + ((tl * COUT) + cbase) * K2 + s * V_ + vv;
#pragma unroll
            for (int r = 0; r < 8; ++r)
                ybp[r * K2] = (__bf16)(acc[r] + cbp[r]);
        }
    }
    __syncthreads();

    // ---- stage 2: 48 16x16 tiles (4 tl x 6 c-tiles x 2 m-tiles), K=96 ----
    for (int t2 = wave; t2 < NT2; t2 += 8) {
        const int tl  = t2 / 12;
        const int rem = t2 % 12;
        const int mt2 = rem / 2, nt2 = rem % 2;
        const __bf16* aBase = yb + ((tl * COUT) + mt2 * 16 + ln) * K2 + hlf * 8;
        const __bf16* bBase = ab + (nt2 * 16 + ln) * K2 + hlf * 8;
        Frag af[3], bfr[3];
#pragma unroll
        for (int kk = 0; kk < 3; ++kk) {
            af[kk].q[0]  = *(const uint4*)(aBase + kk * 32);
            af[kk].q[1]  = *(const uint4*)(aBase + kk * 32 + 16);
            bfr[kk].q[0] = *(const uint4*)(bBase + kk * 32);
            bfr[kk].q[1] = *(const uint4*)(bBase + kk * 32 + 16);
        }
        v8f acc = {};
#pragma unroll
        for (int kk = 0; kk < 3; ++kk)
            acc = __builtin_amdgcn_wmma_f32_16x16x32_bf16(
                false, af[kk].v, false, bfr[kk].v, (short)0, acc, false, false);

        const int m = nt2 * 16 + ln;
        if (m < V_) {
            const int t = t0 + tl;
            const int cb2 = mt2 * 16 + hlf * 8;          // c = cb2 + r, no wrap
            float* zp = z + ((size_t)(b * COUT + cb2) * T_ + t) * V_ + m;
#pragma unroll
            for (int r = 0; r < 8; ++r)
                zp[(size_t)r * (T_ * V_)] = acc[r];      // const 28800B stride
        }
    }
}

// ---------------------------------------------------------------------------
// Kernel C: per-(b,c) slab -> 5-tap temporal filter on the fly -> sum/sumsq
// accumulated with global f32 atomics (no filtered buffer materialized).
// ---------------------------------------------------------------------------
__global__ __launch_bounds__(256) void stats_kernel(const float* __restrict__ z,
                                                    unsigned char* __restrict__ ws) {
    __shared__ float slab[T_ * V_];
    __shared__ float rs[256], rq[256];
    const int tid = threadIdx.x;
    const int c = blockIdx.x % COUT;
    const size_t base = (size_t)blockIdx.x * (T_ * V_);

    for (int i = tid; i < T_ * V_; i += 256) slab[i] = z[base + i];
    __syncthreads();

    float s = 0.0f, q = 0.0f;
    for (int i = tid; i < T_ * V_; i += 256) {
        const int t = i / V_;
        float a = slab[i];
        if (t >= 2)      a += slab[i - 2 * V_];
        if (t >= 1)      a += slab[i - V_];
        if (t <= T_ - 2) a += slab[i + V_];
        if (t <= T_ - 3) a += slab[i + 2 * V_];
        a *= 0.2f;
        s += a; q += a * a;
    }
    rs[tid] = s; rq[tid] = q;
    __syncthreads();
    for (int off = 128; off > 0; off >>= 1) {
        if (tid < off) { rs[tid] += rs[tid + off]; rq[tid] += rq[tid + off]; }
        __syncthreads();
    }
    if (tid == 0) {
        atomicAdd((float*)(ws + WS_SUM)   + c, rs[0]);
        atomicAdd((float*)(ws + WS_SUMSQ) + c, rq[0]);
    }
}

// ---------------------------------------------------------------------------
// Kernel D: fold BN stats into per-channel scale/shift.
// ---------------------------------------------------------------------------
__global__ void bnparam_kernel(const float* __restrict__ gamma,
                               const float* __restrict__ beta,
                               unsigned char* __restrict__ ws) {
    const int c = threadIdx.x;
    if (c < COUT) {
        const float n    = (float)(B_ * T_ * V_);
        const float mean = ((float*)(ws + WS_SUM))[c] / n;
        const float var  = ((float*)(ws + WS_SUMSQ))[c] / n - mean * mean;
        const float inv  = rsqrtf(var + 1e-5f);
        const float sc   = gamma[c] * inv;
        ((float*)(ws + WS_SCALE))[c] = sc;
        ((float*)(ws + WS_SHIFT))[c] = beta[c] - mean * sc;
    }
}

// ---------------------------------------------------------------------------
// Kernel E: recompute filter from z slab (LDS), apply affine + ReLU in place.
// ---------------------------------------------------------------------------
__global__ __launch_bounds__(256) void bnapply_kernel(float* __restrict__ z,
                                                      const unsigned char* __restrict__ ws) {
    __shared__ float slab[T_ * V_];
    const int tid = threadIdx.x;
    const int c = blockIdx.x % COUT;
    const size_t base = (size_t)blockIdx.x * (T_ * V_);
    const float sc = ((const float*)(ws + WS_SCALE))[c];
    const float sh = ((const float*)(ws + WS_SHIFT))[c];

    for (int i = tid; i < T_ * V_; i += 256) slab[i] = z[base + i];
    __syncthreads();
    for (int i = tid; i < T_ * V_; i += 256) {
        const int t = i / V_;
        float a = slab[i];
        if (t >= 2)      a += slab[i - 2 * V_];
        if (t >= 1)      a += slab[i - V_];
        if (t <= T_ - 2) a += slab[i + V_];
        if (t <= T_ - 3) a += slab[i + 2 * V_];
        a = fmaf(a * 0.2f, sc, sh);
        z[base + i] = a > 0.0f ? a : 0.0f;
    }
}

// ---------------------------------------------------------------------------
extern "C" void kernel_launch(void* const* d_in, const int* in_sizes, int n_in,
                              void* d_out, int out_size, void* d_ws, size_t ws_size,
                              hipStream_t stream) {
    const float* x      = (const float*)d_in[0];
    const float* Aadj   = (const float*)d_in[1];
    const float* conv_w = (const float*)d_in[2];
    const float* conv_b = (const float*)d_in[3];
    const float* gamma  = (const float*)d_in[4];
    const float* beta   = (const float*)d_in[5];
    float* out = (float*)d_out;                 // used as z buffer, then final output
    unsigned char* ws = (unsigned char*)d_ws;

    prep_kernel<<<(M_ * K_ / 2 + 255) / 256, 256, 0, stream>>>(conv_w, ws);
    conv_graph_kernel<<<B_ * (T_ / TT), 256, L_TOTAL, stream>>>(x, Aadj, conv_b, ws, out);
    stats_kernel<<<B_ * COUT, 256, 0, stream>>>(out, ws);
    bnparam_kernel<<<1, 128, 0, stream>>>(gamma, beta, ws);
    bnapply_kernel<<<B_ * COUT, 256, 0, stream>>>(out, ws);
}
